// FlowModel_45251775431049
// MI455X (gfx1250) — matrix operand order repaired and checked by reading, not compiled, not run on hardware
//
#include <hip/hip_runtime.h>
#include <math.h>

// ---------------------------------------------------------------------------
// CDNA5 WMMA types
// ---------------------------------------------------------------------------
typedef __attribute__((ext_vector_type(16))) _Float16 v16h;
typedef __attribute__((ext_vector_type(8)))  _Float16 v8h;
typedef __attribute__((ext_vector_type(8)))  float    v8f;

// ---------------------------------------------------------------------------
// Implicit-GEMM conv2d on the WMMA pipe.
//   GEMM view: C[M=Cout, N=B*H*W] = A[M, K=Cin*KSZ*KSZ] * B[K, N]
// Block = 256 threads (8 wave32). Block tile: 32 (M) x 128 (N), K-step 32.
// Wave wv: M tile mt = wv&1 (16 rows), N group ng = wv>>1 (2 x 16 cols)
// -> 16 WMMAs per K-step per block, 2 accumulators per wave.
// KSZ template (1 or 3) turns the K->(ci,kh,kw) decompose into mul/shift.
// act: 0=none 1=relu 2=leaky(0.1) 3=sigmoid
// ---------------------------------------------------------------------------
template<int KSZ>
__launch_bounds__(256)
__global__ void k_conv_wmma(const float* __restrict__ x, const float* __restrict__ w,
                            const float* __restrict__ bias, float* __restrict__ y,
                            int B, int Cin, int H, int W, int Cout,
                            int pad, int dil, int act)
{
    constexpr int KK = KSZ * KSZ;
    __shared__ _Float16 As[32][32];     // [m][kk], row = 64B (16B multiple)
    __shared__ _Float16 Bs[128][40];    // [n][kk], row = 80B (16B multiple)

    const int tid  = threadIdx.x;
    const int wv   = tid >> 5;
    const int lane = tid & 31;
    const int g    = lane >> 4;
    const int l16  = lane & 15;

    const int HW = H * W;
    const int N  = B * HW;
    const int K  = Cin * KK;

    const int nBlock = blockIdx.x * 128;
    const int mBlock = blockIdx.y * 32;

    const int kk = tid & 31;   // fixed K offset within tile (staging)
    const int r0 = tid >> 5;   // base row slot (staging)

    // Loop-invariant spatial decomposition of this thread's 16 B-staging rows.
    int sBB[16], sOH[16], sOW[16];
    #pragma unroll
    for (int t = 0; t < 16; ++t) {
        int gn = nBlock + r0 + 8 * t;
        if (gn < N) {
            int bb = gn / HW, hw = gn - bb * HW;
            int oh = hw / W;
            sBB[t] = bb; sOH[t] = oh; sOW[t] = hw - oh * W;
        } else {
            sBB[t] = -1; sOH[t] = 0; sOW[t] = 0;
        }
    }

    v8f acc0 = {0.f, 0.f, 0.f, 0.f, 0.f, 0.f, 0.f, 0.f};
    v8f acc1 = acc0;

    const int mt = wv & 1;     // wave's M tile
    const int ng = wv >> 1;    // wave's N group (2 tiles)

    for (int k0 = 0; k0 < K; k0 += 32) {
        // hint next weight tile toward the caches (global_prefetch_b8)
        if (tid == 0 && (k0 + 32) < K)
            __builtin_prefetch(w + (size_t)mBlock * K + k0 + 32, 0, 0);

        // One K-decompose per thread per iteration (compile-time constants).
        const int gk = k0 + kk;
        const bool kvalid = (gk < K);
        int ci, kh, kw;
        if (KSZ == 1) { ci = gk; kh = 0; kw = 0; }
        else {
            ci = gk / KK;
            int rem = gk - ci * KK;
            kh = rem / KSZ;
            kw = rem - kh * KSZ;
        }

        // ---- stage A tile: weights [32 x 32] -> LDS f16 ----
        #pragma unroll
        for (int t = 0; t < 4; ++t) {
            int m  = r0 + 8 * t;
            int gm = mBlock + m;
            float v = 0.f;
            if (kvalid && gm < Cout) v = w[(size_t)gm * K + gk];
            As[m][kk] = (_Float16)v;
        }
        // ---- stage B tile: im2col patches [128 n x 32 k] -> LDS f16 ----
        #pragma unroll
        for (int t = 0; t < 16; ++t) {
            float v = 0.f;
            if (kvalid && sBB[t] >= 0) {
                int ih = sOH[t] - pad + kh * dil;
                int iw = sOW[t] - pad + kw * dil;
                if (ih >= 0 && ih < H && iw >= 0 && iw < W)
                    v = x[((size_t)(sBB[t] * Cin + ci) * H + ih) * W + iw];
            }
            Bs[r0 + 8 * t][kk] = (_Float16)v;
        }
        __syncthreads();

        // ---- A fragment (ISA 16-bit 16x32 layout: runs at g*8 and 16+g*8) ----
        const int arow = mt * 16 + l16;
        v8h alo = *(const v8h*)(&As[arow][g * 8]);
        v8h ahi = *(const v8h*)(&As[arow][16 + g * 8]);
        v16h fa;
        #pragma unroll
        for (int j = 0; j < 8; ++j) { fa[j] = alo[j]; fa[8 + j] = ahi[j]; }

        // ---- two B fragments (runs at g*8 and 16+g*8), two WMMAs ----
        #pragma unroll
        for (int q = 0; q < 2; ++q) {
            const int ncol = (ng * 2 + q) * 16 + l16;
            v8h blo = *(const v8h*)(&Bs[ncol][g * 8]);
            v8h bhi = *(const v8h*)(&Bs[ncol][16 + g * 8]);
            v16h fb;
            #pragma unroll
            for (int j = 0; j < 8; ++j) { fb[j] = blo[j]; fb[8 + j] = bhi[j]; }
            if (q == 0)
                acc0 = __builtin_amdgcn_wmma_f32_16x16x32_f16(
                           false, fa, false, fb, (short)0, acc0, false, false);
            else
                acc1 = __builtin_amdgcn_wmma_f32_16x16x32_f16(
                           false, fa, false, fb, (short)0, acc1, false, false);
        }
        __syncthreads();
    }

    // ---- epilogue: C layout M = r + 8*(lane/16), N = lane%16 ----
    #pragma unroll
    for (int q = 0; q < 2; ++q) {
        int gn = nBlock + (ng * 2 + q) * 16 + l16;
        if (gn >= N) continue;
        int bb = gn / HW, hw = gn - bb * HW;
        #pragma unroll
        for (int r = 0; r < 8; ++r) {
            int gm = mBlock + mt * 16 + r + g * 8;
            if (gm < Cout) {
                float v = (q == 0 ? acc0[r] : acc1[r]) + bias[gm];
                if (act == 1)       v = fmaxf(v, 0.f);
                else if (act == 2)  v = (v > 0.f) ? v : 0.1f * v;
                else if (act == 3)  v = 1.f / (1.f + __expf(-v));
                y[((size_t)bb * Cout + gm) * HW + hw] = v;
            }
        }
    }
}

// ---------------------------------------------------------------------------
// Bandwidth-bound helpers
// ---------------------------------------------------------------------------
__global__ void k_cost_volume(const float* __restrict__ f1, const float* __restrict__ f2,
                              float* __restrict__ out, int B, int C, int H, int W, int md)
{
    const int r = 2 * md + 1, S = r * r - 1, HW = H * W;
    int i = blockIdx.x * blockDim.x + threadIdx.x;
    int total = B * S * HW;
    if (i >= total) return;
    int hw = i % HW, s = (i / HW) % S, bb = i / (HW * S);
    int h = hw / W, w0 = hw % W;
    int center = md * r + md;
    int lin = s + (s >= center ? 1 : 0);
    int dy = lin / r - md, dx = lin % r - md;
    int h2 = h + dy, w2 = w0 + dx;
    float acc = 0.f;
    if (h2 >= 0 && h2 < H && w2 >= 0 && w2 < W) {
        const float* p1 = f1 + (size_t)bb * C * HW + hw;
        const float* p2 = f2 + (size_t)bb * C * HW + h2 * W + w2;
        for (int c = 0; c < C; ++c) acc += p1[(size_t)c * HW] * p2[(size_t)c * HW];
    }
    out[i] = acc / (float)C;
}

__global__ void k_warp(const float* __restrict__ x, const float* __restrict__ flow,
                       float* __restrict__ y, int B, int C, int H, int W)
{
    int i = blockIdx.x * blockDim.x + threadIdx.x;
    int HW = H * W, total = B * C * HW;
    if (i >= total) return;
    int hw = i % HW, c = (i / HW) % C, bb = i / (HW * C);
    int h = hw / W, w0 = hw % W;
    float fx = flow[((size_t)bb * 2 + 0) * HW + hw];
    float fy = flow[((size_t)bb * 2 + 1) * HW + hw];
    float px = fminf(fmaxf((float)w0 + fx, 0.f), (float)(W - 1));
    float py = fminf(fmaxf((float)h  + fy, 0.f), (float)(H - 1));
    int x0 = (int)floorf(px), y0 = (int)floorf(py);
    float wx = px - x0, wy = py - y0;
    int x1 = min(x0 + 1, W - 1), y1 = min(y0 + 1, H - 1);
    const float* img = x + ((size_t)bb * C + c) * HW;
    float Ia = img[y0 * W + x0], Ib = img[y0 * W + x1];
    float Ic = img[y1 * W + x0], Id = img[y1 * W + x1];
    y[i] = Ia * (1 - wx) * (1 - wy) + Ib * wx * (1 - wy)
         + Ic * (1 - wx) * wy + Id * wx * wy;
}

// align-corners bilinear resize with fused scalar multiply
__global__ void k_up_ac(const float* __restrict__ x, float* __restrict__ y,
                        int B, int C, int H, int W, int OH, int OW, float scale)
{
    int i = blockIdx.x * blockDim.x + threadIdx.x;
    int OHW = OH * OW, total = B * C * OHW;
    if (i >= total) return;
    int ohw = i % OHW, c = (i / OHW) % C, bb = i / (OHW * C);
    int oh = ohw / OW, ow = ohw % OW;
    float ry = (OH > 1) ? (float)(H - 1) / (float)(OH - 1) : 0.f;
    float rx = (OW > 1) ? (float)(W - 1) / (float)(OW - 1) : 0.f;
    float sy = oh * ry, sx = ow * rx;
    int y0 = (int)floorf(sy), x0 = (int)floorf(sx);
    int y1 = min(y0 + 1, H - 1), x1 = min(x0 + 1, W - 1);
    float wy = sy - y0, wx = sx - x0;
    const float* img = x + ((size_t)bb * C + c) * H * W;
    float t0 = img[y0 * W + x0] * (1 - wx) + img[y0 * W + x1] * wx;
    float t1 = img[y1 * W + x0] * (1 - wx) + img[y1 * W + x1] * wx;
    y[i] = (t0 * (1 - wy) + t1 * wy) * scale;
}

// channel-concat copy: dst[b, c0+c, :, :] = src[b, c, :, :]
__global__ void k_chcopy(const float* __restrict__ src, float* __restrict__ dst,
                         int Cs, int Ct, int c0, int HW, int n)
{
    int i = blockIdx.x * blockDim.x + threadIdx.x;
    if (i >= n) return;
    int hw = i % HW, c = (i / HW) % Cs, bb = i / (HW * Cs);
    dst[((size_t)bb * Ct + c0 + c) * HW + hw] = src[i];
}

// out = e*ef + (1-e)*sp   (e is [B,1,H,W], broadcast over C)
__global__ void k_eac_combine(const float* __restrict__ e, const float* __restrict__ ef,
                              const float* __restrict__ sp, float* __restrict__ out,
                              int C, int HW, int n)
{
    int i = blockIdx.x * blockDim.x + threadIdx.x;
    if (i >= n) return;
    int hw = i % HW, bb = i / (HW * C);
    float ev = e[(size_t)bb * HW + hw];
    out[i] = ev * ef[i] + (1.f - ev) * sp[i];
}

__global__ void k_relu_add(const float* __restrict__ a, const float* __restrict__ b,
                           float* __restrict__ out, int n)
{
    int i = blockIdx.x * blockDim.x + threadIdx.x;
    if (i < n) out[i] = fmaxf(a[i], 0.f) + b[i];
}
__global__ void k_relu(float* __restrict__ x, int n)
{
    int i = blockIdx.x * blockDim.x + threadIdx.x;
    if (i < n) x[i] = fmaxf(x[i], 0.f);
}
__global__ void k_add(const float* __restrict__ a, const float* __restrict__ b,
                      float* __restrict__ out, int n)
{
    int i = blockIdx.x * blockDim.x + threadIdx.x;
    if (i < n) out[i] = a[i] + b[i];
}
__global__ void k_mul(const float* __restrict__ a, const float* __restrict__ b,
                      float* __restrict__ out, int n)
{
    int i = blockIdx.x * blockDim.x + threadIdx.x;
    if (i < n) out[i] = a[i] * b[i];
}

// global mean+max pool over H*W, one block per (b,c)
__global__ void k_pool_meanmax(const float* __restrict__ x, float* __restrict__ avg,
                               float* __restrict__ mx, int HW)
{
    int bc = blockIdx.x;
    const float* p = x + (size_t)bc * HW;
    float s = 0.f, m = -3.0e38f;
    for (int i = threadIdx.x; i < HW; i += blockDim.x) {
        float v = p[i]; s += v; m = fmaxf(m, v);
    }
    __shared__ float ss[256], sm[256];
    ss[threadIdx.x] = s; sm[threadIdx.x] = m;
    __syncthreads();
    for (int o = 128; o > 0; o >>= 1) {
        if ((int)threadIdx.x < o) {
            ss[threadIdx.x] += ss[threadIdx.x + o];
            sm[threadIdx.x] = fmaxf(sm[threadIdx.x], sm[threadIdx.x + o]);
        }
        __syncthreads();
    }
    if (threadIdx.x == 0) { avg[bc] = ss[0] / (float)HW; mx[bc] = sm[0]; }
}

// channel attention: hidden = relu(W1*v + b1) for v in {avg, max}
__global__ void k_ca_hidden(const float* __restrict__ avg, const float* __restrict__ mx,
                            const float* __restrict__ w1, const float* __restrict__ b1,
                            float* __restrict__ hidA, float* __restrict__ hidM,
                            int B, int C, int Hd)
{
    int i = blockIdx.x * blockDim.x + threadIdx.x;
    if (i >= B * Hd) return;
    int bb = i / Hd, h = i % Hd;
    float sa = b1[h], sm2 = b1[h];
    for (int c = 0; c < C; ++c) {
        float ww = w1[h * C + c];
        sa  += ww * avg[bb * C + c];
        sm2 += ww * mx[bb * C + c];
    }
    hidA[i] = fmaxf(sa, 0.f);
    hidM[i] = fmaxf(sm2, 0.f);
}
// attn = sigmoid( W2*hidA + b2 + W2*hidM + b2 )
__global__ void k_ca_attn(const float* __restrict__ hidA, const float* __restrict__ hidM,
                          const float* __restrict__ w2, const float* __restrict__ b2,
                          float* __restrict__ attn, int B, int C, int Hd)
{
    int i = blockIdx.x * blockDim.x + threadIdx.x;
    if (i >= B * C) return;
    int bb = i / C, c = i % C;
    float s = 2.f * b2[c];
    for (int h = 0; h < Hd; ++h)
        s += w2[c * Hd + h] * (hidA[bb * Hd + h] + hidM[bb * Hd + h]);
    attn[i] = 1.f / (1.f + __expf(-s));
}
__global__ void k_scale_ch(float* __restrict__ x, const float* __restrict__ attn,
                           int C, int HW, int n)
{
    int i = blockIdx.x * blockDim.x + threadIdx.x;
    if (i >= n) return;
    int c = (i / HW) % C, bb = i / (HW * C);
    x[i] *= attn[bb * C + c];
}

// ---------------------------------------------------------------------------
// Host-side orchestration
// ---------------------------------------------------------------------------
struct Conv { const float* w; const float* b; int cin, cout, kh, kw; };
struct EacP { Conv s, e, ed0, ed1, ed2, ed3; };
struct FebP { EacP c1, c2; Conv c3, sk; };
struct CtxP { Conv convs[7]; const float *f1w, *f1b, *f2w, *f2b; };

struct Arena {
    char* base; size_t off, cap;
    float* alloc(size_t nflt) {
        size_t bytes = ((nflt * sizeof(float)) + 255) & ~(size_t)255;
        float* r = (float*)(base + off);
        off += bytes;
        return r;
    }
};

static inline dim3 G1(int n) { return dim3((n + 255) / 256); }

static void conv2d(hipStream_t st, const float* x, const Conv& p, float* y,
                   int B, int H, int W, int pad, int dil, int act)
{
    int N = B * H * W;
    dim3 grid((N + 127) / 128, (p.cout + 31) / 32);
    if (p.kh == 1)
        k_conv_wmma<1><<<grid, 256, 0, st>>>(x, p.w, p.b, y, B, p.cin, H, W,
                                             p.cout, pad, dil, act);
    else
        k_conv_wmma<3><<<grid, 256, 0, st>>>(x, p.w, p.b, y, B, p.cin, H, W,
                                             p.cout, pad, dil, act);
}

// eac: returns combined output (out, Cout ch) and edge map (e_out, 1 ch)
static void eac(hipStream_t st, Arena& A, const float* x, const EacP& p,
                int B, int H, int W, float* out, float* e_out)
{
    size_t mark = A.off;
    int HW = H * W;
    float* t32a = A.alloc((size_t)B * 32 * HW);
    float* t64  = A.alloc((size_t)B * 64 * HW);
    float* t32b = A.alloc((size_t)B * 32 * HW);
    float* sp   = A.alloc((size_t)B * p.s.cout * HW);
    float* ef   = A.alloc((size_t)B * p.s.cout * HW);
    conv2d(st, x,    p.ed0, t32a, B, H, W, 1, 1, 1);
    conv2d(st, t32a, p.ed1, t64,  B, H, W, 1, 1, 1);
    conv2d(st, t64,  p.ed2, t32b, B, H, W, 1, 1, 1);
    conv2d(st, t32b, p.ed3, e_out, B, H, W, 1, 1, 3);   // sigmoid
    conv2d(st, x, p.s, sp, B, H, W, p.s.kh / 2, 1, 0);  // spatial conv
    conv2d(st, x, p.e, ef, B, H, W, 0, 1, 0);           // 1x1 edge conv
    int n = B * p.s.cout * HW;
    k_eac_combine<<<G1(n), 256, 0, st>>>(e_out, ef, sp, out, p.s.cout, HW, n);
    A.off = mark;
}

static void feb(hipStream_t st, Arena& A, const float* x, const FebP& p,
                int B, int H, int W, float* flow_out, float* e_out)
{
    size_t mark = A.off;
    int HW = H * W, hid = p.c1.s.cout;
    int n = B * hid * HW;
    float* idt = A.alloc((size_t)n);
    float* y1  = A.alloc((size_t)n);
    float* y2  = A.alloc((size_t)n);
    float* e1  = A.alloc((size_t)B * HW);
    float* e2  = A.alloc((size_t)B * HW);
    conv2d(st, x, p.sk, idt, B, H, W, 0, 1, 0);
    eac(st, A, x, p.c1, B, H, W, y1, e1);
    k_relu_add<<<G1(n), 256, 0, st>>>(y1, idt, y1, n);
    eac(st, A, y1, p.c2, B, H, W, y2, e2);
    k_relu_add<<<G1(n), 256, 0, st>>>(y2, y1, y2, n);
    conv2d(st, y2, p.c3, flow_out, B, H, W, 1, 1, 0);
    k_mul<<<G1(B * HW), 256, 0, st>>>(e1, e2, e_out, B * HW);
    A.off = mark;
}

static void ctx_net(hipStream_t st, Arena& A, const float* xin, const CtxP& p,
                    int B, int H, int W, float* out)
{
    size_t mark = A.off;
    int HW = H * W;
    float* xa = A.alloc((size_t)B * 128 * HW);
    float* xb = A.alloc((size_t)B * 128 * HW);
    conv2d(st, xin, p.convs[0], xa, B, H, W, 1, 1, 2);   // leaky 0.1
    // channel attention
    float* avg  = A.alloc((size_t)B * 128);
    float* mx   = A.alloc((size_t)B * 128);
    float* hA   = A.alloc((size_t)B * 8);
    float* hM   = A.alloc((size_t)B * 8);
    float* attn = A.alloc((size_t)B * 128);
    k_pool_meanmax<<<B * 128, 256, 0, st>>>(xa, avg, mx, HW);
    k_ca_hidden<<<1, 64, 0, st>>>(avg, mx, p.f1w, p.f1b, hA, hM, B, 128, 8);
    k_ca_attn<<<G1(B * 128), 256, 0, st>>>(hA, hM, p.f2w, p.f2b, attn, B, 128, 8);
    int nx = B * 128 * HW;
    k_scale_ch<<<G1(nx), 256, 0, st>>>(xa, attn, 128, HW, nx);
    // dilated tower
    const int dils[5] = {2, 4, 8, 16, 1};
    float* cur = xa; float* nxt = xb;
    for (int i = 0; i < 5; ++i) {
        conv2d(st, cur, p.convs[1 + i], nxt, B, H, W, dils[i], dils[i], 2);
        float* t = cur; cur = nxt; nxt = t;
    }
    conv2d(st, cur, p.convs[6], out, B, H, W, 1, 1, 0);
    A.off = mark;
}

extern "C" void kernel_launch(void* const* d_in, const int* in_sizes, int n_in,
                              void* d_out, int out_size, void* d_ws, size_t ws_size,
                              hipStream_t stream)
{
    (void)in_sizes; (void)n_in; (void)out_size;
    const int B = 4;

    // ---- unpack parameters (flattened in make_params() insertion order) ----
    int pi = 7;
    auto C_ = [&](int cout, int cin, int k) -> Conv {
        Conv c{ (const float*)d_in[pi], (const float*)d_in[pi + 1], cin, cout, k, k };
        pi += 2; return c;
    };
    auto E_ = [&](int cin, int cout) -> EacP {
        EacP e;
        e.s   = C_(cout, cin, 3);
        e.e   = C_(cout, cin, 1);
        e.ed0 = C_(32, cin, 3);
        e.ed1 = C_(64, 32, 3);
        e.ed2 = C_(32, 64, 3);
        e.ed3 = C_(1, 32, 3);
        return e;
    };
    auto F_ = [&](int cin, int hid) -> FebP {
        FebP f;
        f.c1 = E_(cin, hid);
        f.c2 = E_(hid, hid);
        f.c3 = C_(2, hid, 3);
        f.sk = C_(hid, cin, 1);
        return f;
    };
    Conv conv_s1 = C_(256, 512, 1);
    Conv conv_s2 = C_(128, 256, 1);
    Conv conv_s3 = C_(64, 128, 1);
    FebP fs1 = F_(592, 128);      // 256*2 + 80
    FebP fs2 = F_(338, 96);       // 128*2 + 80 + 2
    FebP fs3 = F_(210, 64);       // 64*2 + 80 + 2
    CtxP ctx;
    ctx.convs[0] = C_(128, 130, 3);
    ctx.convs[1] = C_(128, 128, 3);
    ctx.convs[2] = C_(128, 128, 3);
    ctx.convs[3] = C_(96, 128, 3);
    ctx.convs[4] = C_(64, 96, 3);
    ctx.convs[5] = C_(32, 64, 3);
    ctx.convs[6] = C_(2, 32, 3);
    ctx.f1w = (const float*)d_in[pi++]; ctx.f1b = (const float*)d_in[pi++];
    ctx.f2w = (const float*)d_in[pi++]; ctx.f2b = (const float*)d_in[pi++];
    EacP r0 = E_(5, 32);
    EacP r2 = E_(32, 32);
    Conv r4 = C_(2, 32, 3);

    const float* img1  = (const float*)d_in[0];
    const float* f1_s1 = (const float*)d_in[1];
    const float* f2_s1 = (const float*)d_in[2];
    const float* f1_s2 = (const float*)d_in[3];
    const float* f2_s2 = (const float*)d_in[4];
    const float* f1_s3 = (const float*)d_in[5];
    const float* f2_s3 = (const float*)d_in[6];

    Arena A{ (char*)d_ws, 0, ws_size };
    hipStream_t st = stream;

    // ---- feature reduction (1x1 convs on WMMA pipe) ----
    const int HW1 = 400, HW2 = 1600, HW3 = 6400;
    float* f1s1 = A.alloc((size_t)B * 256 * HW1);
    float* f2s1 = A.alloc((size_t)B * 256 * HW1);
    float* f1s2 = A.alloc((size_t)B * 128 * HW2);
    float* f2s2 = A.alloc((size_t)B * 128 * HW2);
    float* f1s3 = A.alloc((size_t)B * 64 * HW3);
    float* f2s3 = A.alloc((size_t)B * 64 * HW3);
    conv2d(st, f1_s1, conv_s1, f1s1, B, 20, 20, 0, 1, 0);
    conv2d(st, f2_s1, conv_s1, f2s1, B, 20, 20, 0, 1, 0);
    conv2d(st, f1_s2, conv_s2, f1s2, B, 40, 40, 0, 1, 0);
    conv2d(st, f2_s2, conv_s2, f2s2, B, 40, 40, 0, 1, 0);
    conv2d(st, f1_s3, conv_s3, f1s3, B, 80, 80, 0, 1, 0);
    conv2d(st, f2_s3, conv_s3, f2s3, B, 80, 80, 0, 1, 0);

    // ---- scale 1 (20x20) ----
    float* cv1 = A.alloc((size_t)B * 80 * HW1);
    k_cost_volume<<<G1(B * 80 * HW1), 256, 0, st>>>(f1s1, f2s1, cv1, B, 256, 20, 20, 4);
    float* cat1 = A.alloc((size_t)B * 592 * HW1);
    k_chcopy<<<G1(B * 256 * HW1), 256, 0, st>>>(f1s1, cat1, 256, 592, 0,   HW1, B * 256 * HW1);
    k_chcopy<<<G1(B * 256 * HW1), 256, 0, st>>>(f2s1, cat1, 256, 592, 256, HW1, B * 256 * HW1);
    k_chcopy<<<G1(B * 80  * HW1), 256, 0, st>>>(cv1,  cat1, 80,  592, 512, HW1, B * 80 * HW1);
    float* flow1 = A.alloc((size_t)B * 2 * HW1);
    float* e1    = A.alloc((size_t)B * HW1);
    feb(st, A, cat1, fs1, B, 20, 20, flow1, e1);
    float* flow1_up = A.alloc((size_t)B * 2 * HW2);
    float* e1_up    = A.alloc((size_t)B * HW2);
    k_up_ac<<<G1(B * 2 * HW2), 256, 0, st>>>(flow1, flow1_up, B, 2, 20, 20, 40, 40, 2.f);
    k_up_ac<<<G1(B * HW2),     256, 0, st>>>(e1,    e1_up,    B, 1, 20, 20, 40, 40, 1.f);

    // ---- scale 2 (40x40) ----
    float* wf2s2 = A.alloc((size_t)B * 128 * HW2);
    k_warp<<<G1(B * 128 * HW2), 256, 0, st>>>(f2s2, flow1_up, wf2s2, B, 128, 40, 40);
    float* cv2 = A.alloc((size_t)B * 80 * HW2);
    k_cost_volume<<<G1(B * 80 * HW2), 256, 0, st>>>(f1s2, wf2s2, cv2, B, 128, 40, 40, 4);
    float* cat2 = A.alloc((size_t)B * 338 * HW2);
    k_chcopy<<<G1(B * 128 * HW2), 256, 0, st>>>(f1s2,     cat2, 128, 338, 0,   HW2, B * 128 * HW2);
    k_chcopy<<<G1(B * 128 * HW2), 256, 0, st>>>(wf2s2,    cat2, 128, 338, 128, HW2, B * 128 * HW2);
    k_chcopy<<<G1(B * 80  * HW2), 256, 0, st>>>(cv2,      cat2, 80,  338, 256, HW2, B * 80 * HW2);
    k_chcopy<<<G1(B * 2   * HW2), 256, 0, st>>>(flow1_up, cat2, 2,   338, 336, HW2, B * 2 * HW2);
    float* d2 = A.alloc((size_t)B * 2 * HW2);
    float* e2 = A.alloc((size_t)B * HW2);
    feb(st, A, cat2, fs2, B, 40, 40, d2, e2);
    float* flow2 = A.alloc((size_t)B * 2 * HW2);
    float* ec2   = A.alloc((size_t)B * HW2);
    k_add<<<G1(B * 2 * HW2), 256, 0, st>>>(flow1_up, d2, flow2, B * 2 * HW2);
    k_mul<<<G1(B * HW2),     256, 0, st>>>(e1_up, e2, ec2, B * HW2);
    float* flow2_up = A.alloc((size_t)B * 2 * HW3);
    float* e2_up    = A.alloc((size_t)B * HW3);
    k_up_ac<<<G1(B * 2 * HW3), 256, 0, st>>>(flow2, flow2_up, B, 2, 40, 40, 80, 80, 2.f);
    k_up_ac<<<G1(B * HW3),     256, 0, st>>>(ec2,   e2_up,    B, 1, 40, 40, 80, 80, 1.f);

    // ---- scale 3 (80x80) + context ----
    float* wf2s3 = A.alloc((size_t)B * 64 * HW3);
    k_warp<<<G1(B * 64 * HW3), 256, 0, st>>>(f2s3, flow2_up, wf2s3, B, 64, 80, 80);
    float* cv3 = A.alloc((size_t)B * 80 * HW3);
    k_cost_volume<<<G1(B * 80 * HW3), 256, 0, st>>>(f1s3, wf2s3, cv3, B, 64, 80, 80, 4);
    float* cat3 = A.alloc((size_t)B * 210 * HW3);
    k_chcopy<<<G1(B * 64 * HW3), 256, 0, st>>>(f1s3,     cat3, 64, 210, 0,   HW3, B * 64 * HW3);
    k_chcopy<<<G1(B * 64 * HW3), 256, 0, st>>>(wf2s3,    cat3, 64, 210, 64,  HW3, B * 64 * HW3);
    k_chcopy<<<G1(B * 80 * HW3), 256, 0, st>>>(cv3,      cat3, 80, 210, 128, HW3, B * 80 * HW3);
    k_chcopy<<<G1(B * 2  * HW3), 256, 0, st>>>(flow2_up, cat3, 2,  210, 208, HW3, B * 2 * HW3);
    float* d3 = A.alloc((size_t)B * 2 * HW3);
    float* e3 = A.alloc((size_t)B * HW3);
    feb(st, A, cat3, fs3, B, 80, 80, d3, e3);
    float* flow3 = A.alloc((size_t)B * 2 * HW3);
    k_add<<<G1(B * 2 * HW3), 256, 0, st>>>(flow2_up, d3, flow3, B * 2 * HW3);
    float* catc = A.alloc((size_t)B * 130 * HW3);
    k_chcopy<<<G1(B * 64 * HW3), 256, 0, st>>>(f1s3,  catc, 64, 130, 0,   HW3, B * 64 * HW3);
    k_chcopy<<<G1(B * 64 * HW3), 256, 0, st>>>(wf2s3, catc, 64, 130, 64,  HW3, B * 64 * HW3);
    k_chcopy<<<G1(B * 2  * HW3), 256, 0, st>>>(flow3, catc, 2,  130, 128, HW3, B * 2 * HW3);
    float* ctxo = A.alloc((size_t)B * 2 * HW3);
    ctx_net(st, A, catc, ctx, B, 80, 80, ctxo);
    float* flow3r = A.alloc((size_t)B * 2 * HW3);
    float* ec3    = A.alloc((size_t)B * HW3);
    k_add<<<G1(B * 2 * HW3), 256, 0, st>>>(flow3, ctxo, flow3r, B * 2 * HW3);
    k_mul<<<G1(B * HW3),     256, 0, st>>>(e2_up, e3, ec3, B * HW3);

    // ---- refinement at 160x160 ----
    const int HW4 = 160 * 160;
    float* flow160 = A.alloc((size_t)B * 2 * HW4);
    float* edge160 = A.alloc((size_t)B * HW4);
    float* img1r   = A.alloc((size_t)B * 3 * HW4);
    k_up_ac<<<G1(B * 2 * HW4), 256, 0, st>>>(flow3r, flow160, B, 2, 80, 80, 160, 160, 2.f);
    k_up_ac<<<G1(B * HW4),     256, 0, st>>>(ec3,    edge160, B, 1, 80, 80, 160, 160, 1.f);
    k_up_ac<<<G1(B * 3 * HW4), 256, 0, st>>>(img1,   img1r,   B, 3, 640, 640, 160, 160, 1.f);
    float* cat5 = A.alloc((size_t)B * 5 * HW4);
    k_chcopy<<<G1(B * 2 * HW4), 256, 0, st>>>(flow160, cat5, 2, 5, 0, HW4, B * 2 * HW4);
    k_chcopy<<<G1(B * 3 * HW4), 256, 0, st>>>(img1r,   cat5, 3, 5, 2, HW4, B * 3 * HW4);
    float* x32  = A.alloc((size_t)B * 32 * HW4);
    float* re   = A.alloc((size_t)B * HW4);
    float* x32b = A.alloc((size_t)B * 32 * HW4);
    float* etmp = A.alloc((size_t)B * HW4);
    eac(st, A, cat5, r0, B, 160, 160, x32, re);
    k_relu<<<G1(B * 32 * HW4), 256, 0, st>>>(x32, B * 32 * HW4);
    eac(st, A, x32, r2, B, 160, 160, x32b, etmp);
    k_relu<<<G1(B * 32 * HW4), 256, 0, st>>>(x32b, B * 32 * HW4);
    float* dflow = A.alloc((size_t)B * 2 * HW4);
    conv2d(st, x32b, r4, dflow, B, 160, 160, 1, 1, 0);
    float* fedge = A.alloc((size_t)B * HW4);
    k_mul<<<G1(B * HW4), 256, 0, st>>>(edge160, re, fedge, B * HW4);

    // ---- outputs: rflow160 | flow640 | edge640 ----
    float* out0 = (float*)d_out;               // [4,2,160,160]
    float* out1 = out0 + (size_t)B * 2 * HW4;  // [4,2,640,640]
    float* out2 = out1 + (size_t)B * 2 * 640 * 640;  // [4,1,640,640]
    k_add<<<G1(B * 2 * HW4), 256, 0, st>>>(flow160, dflow, out0, B * 2 * HW4);
    k_up_ac<<<G1(B * 2 * 640 * 640), 256, 0, st>>>(out0,  out1, B, 2, 160, 160, 640, 640, 4.f);
    k_up_ac<<<G1(B * 1 * 640 * 640), 256, 0, st>>>(fedge, out2, B, 1, 160, 160, 640, 640, 1.f);
}
